// MambaModel_73890617360478
// MI455X (gfx1250) — matrix operand order repaired and checked by reading, not compiled, not run on hardware
//
#include <hip/hip_runtime.h>
#include <hip/hip_bf16.h>
#include <math.h>

typedef __attribute__((ext_vector_type(16))) _Float16 v16h;
typedef __attribute__((ext_vector_type(8)))  _Float16 v8h;
typedef __attribute__((ext_vector_type(8)))  float    v8f;
typedef __attribute__((ext_vector_type(4)))  unsigned int v4u;
typedef __attribute__((ext_vector_type(8)))  int      v8i;
typedef __attribute__((ext_vector_type(4)))  int      v4i;

#define T_TOKENS 36864   // 4096 * 9
#define NBATCH   4096
#define SEQ      9
#define DMODEL   256
#define DINNER   512
#define NSTATE   16

static __device__ __forceinline__ float silu_f(float x) { return x / (1.0f + expf(-x)); }

// ---------------------------------------------------------------------------
// TDM 2D tile load: global (row-major, fp32, row stride stride_d0 elements)
// -> LDS at byte offset lds_off. Descriptor packing per CDNA5 ISA ch.8.
// Issued once per calling wave; caller must s_wait_tensorcnt + barrier.
// This toolchain exposes the 6-arg builtin: (g0, g1, g2, g3, g4, cpol).
// ---------------------------------------------------------------------------
static __device__ __forceinline__ void tdm_load_2d_f32(
    unsigned lds_off, const float* gaddr,
    unsigned tile_d0, unsigned tile_d1,
    unsigned tensor_d0, unsigned tensor_d1, unsigned stride_d0) {
  unsigned long long ga = (unsigned long long)(uintptr_t)gaddr;
  v4u g0;
  g0[0] = 1u;                                   // count=1, user-mode descriptor
  g0[1] = lds_off;                              // LDS byte address
  g0[2] = (unsigned)(ga & 0xFFFFFFFFu);         // global_addr[31:0]
  g0[3] = (unsigned)((ga >> 32) & 0x01FFFFFFu)  // global_addr[56:32]
          | 0x80000000u;                        // type = 2 ("image")
  v8i g1;
  g1[0] = (int)(2u << 16);                      // data_size = 2 (4 bytes)
  g1[1] = (int)((tensor_d0 & 0xFFFFu) << 16);   // tensor_dim0[15:0]
  g1[2] = (int)((tensor_d0 >> 16) | ((tensor_d1 & 0xFFFFu) << 16));
  g1[3] = (int)((tensor_d1 >> 16) | (tile_d0 << 16));   // tile_dim0
  g1[4] = (int)(tile_d1 & 0xFFFFu);             // tile_dim1 (tile_dim2 = 0)
  g1[5] = (int)stride_d0;                       // tensor_dim0_stride[31:0]
  g1[6] = 0;                                    // stride[47:32] | dim1_stride lo
  g1[7] = 0;
  v4i z4 = {0, 0, 0, 0};
  v8i z8 = {0, 0, 0, 0, 0, 0, 0, 0};
  __builtin_amdgcn_tensor_load_to_lds(g0, g1, z4, z4, z8, 0);
}

// ---------------------------------------------------------------------------
// Generic fp32 -> f16 conversion (used once per weight tensor at launch start)
// ---------------------------------------------------------------------------
__global__ void k_cvt16(const float* __restrict__ src, _Float16* __restrict__ dst, int n) {
  int id = blockIdx.x * blockDim.x + threadIdx.x;
  if (id < n) dst[id] = (_Float16)src[id];
}

// ---------------------------------------------------------------------------
// h[t][j] = x[t] * w1[j] + b1[j]   (dual f32 + f16 output)
// ---------------------------------------------------------------------------
__global__ void k_embed(const float* __restrict__ x, const float* __restrict__ w1,
                        const float* __restrict__ b1, float* __restrict__ h,
                        _Float16* __restrict__ h16) {
  int id = blockIdx.x * blockDim.x + threadIdx.x;
  if (id >= T_TOKENS * DMODEL) return;
  int t = id >> 8, j = id & 255;
  float v = x[t] * w1[j] + b1[j];
  h[id] = v;
  h16[id] = (_Float16)v;
}

// ---------------------------------------------------------------------------
// C[M x N] = act(A16[M x K] @ W16[N x K]^T + bias); f16 in, f32 out (+f16 mirror).
// One wave computes a 16 x (16*TN) strip. Per K-chunk: load A fragment once,
// load all TN B fragments, then issue TN WMMAs (keeps loads in flight).
// ---------------------------------------------------------------------------
template <int TN>
__global__ void k_gemm_wmma_f16(const _Float16* __restrict__ A,
                                const _Float16* __restrict__ W,
                                const float* __restrict__ bias,
                                float* __restrict__ C, _Float16* __restrict__ C16,
                                int M, int N, int K, int relu) {
  const int lane = threadIdx.x & 31;
  const int wave = threadIdx.x >> 5;
  const int colGroups = N / (16 * TN);
  const int totalGroups = (M >> 4) * colGroups;
  const int grp = blockIdx.x * (blockDim.x >> 5) + wave;
  if (grp >= totalGroups) return;
  const int row0 = (grp / colGroups) << 4;
  const int col0 = (grp % colGroups) * (16 * TN);

  const int nrow = lane & 15;   // A row (m) and B col (n) owned by this lane
  const int hi   = lane >> 4;

  const _Float16* arow = A + (size_t)(row0 + nrow) * K;
  const _Float16* wrow[TN];
#pragma unroll
  for (int j = 0; j < TN; ++j)
    wrow[j] = W + (size_t)(col0 + 16 * j + nrow) * K;

  v8f acc[TN];
#pragma unroll
  for (int j = 0; j < TN; ++j) acc[j] = (v8f){};

  for (int k0 = 0; k0 < K; k0 += 32) {
    // A fragment: halves j<8 -> k = k0+8*hi+j ; j>=8 -> k = k0+16+8*hi+(j-8)
    v8h a_lo = *(const v8h*)(arow + k0 + 8 * hi);
    v8h a_hi = *(const v8h*)(arow + k0 + 16 + 8 * hi);
    v16h af = __builtin_shufflevector(a_lo, a_hi, 0, 1, 2, 3, 4, 5, 6, 7,
                                      8, 9, 10, 11, 12, 13, 14, 15);
    if (k0 + 32 < K) __builtin_prefetch(arow + k0 + 32, 0, 3);
    // Load all B fragments first, then issue the WMMAs.
    v16h bfr[TN];
#pragma unroll
    for (int j = 0; j < TN; ++j) {
      const _Float16* pb = wrow[j] + k0 + 8 * hi;   // k = k0+8*hi+(0..7), k0+16+8*hi+(0..7)
      v8h b_lo = *(const v8h*)(pb);
      v8h b_hi = *(const v8h*)(pb + 16);
      bfr[j] = __builtin_shufflevector(b_lo, b_hi, 0, 1, 2, 3, 4, 5, 6, 7,
                                       8, 9, 10, 11, 12, 13, 14, 15);
    }
#pragma unroll
    for (int j = 0; j < TN; ++j) {
      acc[j] = __builtin_amdgcn_wmma_f32_16x16x32_f16(
          /*neg_a=*/false, af, /*neg_b=*/false, bfr[j],
          /*c_mod=*/(short)0, acc[j], /*reuse_a=*/false, /*reuse_b=*/false);
    }
  }

#pragma unroll
  for (int j = 0; j < TN; ++j) {
    const int cn = col0 + 16 * j + nrow;
    const float bb = bias ? bias[cn] : 0.0f;
#pragma unroll
    for (int r = 0; r < 8; ++r) {
      int m = r + 8 * hi;                 // C row for VGPR r
      float v = acc[j][r] + bb;
      if (relu) v = fmaxf(v, 0.0f);
      size_t idx = (size_t)(row0 + m) * N + cn;
      C[idx] = v;
      if (C16) C16[idx] = (_Float16)v;
    }
  }
}

// ---------------------------------------------------------------------------
// Attention for one (batch, head): S=9, hd=128. qkv row stride = 768.
// q/k/v tiles staged into LDS via the Tensor Data Mover (3 x 2D descriptors),
// then scores/softmax/AV in fp32. Dual f32 + f16 output.
// Dynamic LDS layout (bytes): q @ 0, k @ 4608, v @ 9216, scores @ 13824.
// ---------------------------------------------------------------------------
__global__ void k_attn(const float* __restrict__ qkv, float* __restrict__ outb,
                       _Float16* __restrict__ out16) {
  extern __shared__ float smem[];
  float* sq = smem;               // 9*128
  float* sk = smem + 1152;        // 9*128
  float* sv = smem + 2304;        // 9*128
  float* ss = smem + 3456;        // 9*9
  const int b = blockIdx.x, hh = blockIdx.y;
  const int tid = threadIdx.x;    // 0..127

  if (tid < 32) {   // wave 0 issues the three TDM tile loads
    const float* base = qkv + (size_t)(b * SEQ) * 768 + hh * 128;
    tdm_load_2d_f32(/*lds_off=*/0,    base,       128, SEQ, 768, T_TOKENS, 768);
    tdm_load_2d_f32(/*lds_off=*/4608, base + 256, 128, SEQ, 768, T_TOKENS, 768);
    tdm_load_2d_f32(/*lds_off=*/9216, base + 512, 128, SEQ, 768, T_TOKENS, 768);
    __builtin_amdgcn_s_wait_tensorcnt(0);
  }
  __syncthreads();

  if (tid < SEQ * SEQ) {
    int iq = tid / SEQ, ik = tid % SEQ;
    float acc = 0.0f;
    for (int dd = 0; dd < 128; ++dd) acc += sq[iq * 128 + dd] * sk[ik * 128 + dd];
    ss[iq * SEQ + ik] = acc * 0.08838834764831845f;   // 1/sqrt(128)
  }
  __syncthreads();
  if (tid < SEQ) {
    float mx = -1e30f;
    for (int j = 0; j < SEQ; ++j) mx = fmaxf(mx, ss[tid * SEQ + j]);
    float e[SEQ], sum = 0.0f;
    for (int j = 0; j < SEQ; ++j) { e[j] = expf(ss[tid * SEQ + j] - mx); sum += e[j]; }
    float inv = 1.0f / sum;
    for (int j = 0; j < SEQ; ++j) ss[tid * SEQ + j] = e[j] * inv;
  }
  __syncthreads();
  for (int iq = 0; iq < SEQ; ++iq) {
    float acc = 0.0f;
    for (int ik = 0; ik < SEQ; ++ik) acc += ss[iq * SEQ + ik] * sv[ik * 128 + tid];
    size_t idx = (size_t)(b * SEQ + iq) * 256 + hh * 128 + tid;
    outb[idx] = acc;
    out16[idx] = (_Float16)acc;
  }
}

// ---------------------------------------------------------------------------
// h[t] = LayerNorm(h[t] + r[t]) over d=256; dual f32 + f16 output.
// ---------------------------------------------------------------------------
__global__ void k_add_ln(float* __restrict__ h, const float* __restrict__ r,
                         const float* __restrict__ g, const float* __restrict__ be,
                         _Float16* __restrict__ h16) {
  const int t = blockIdx.x, j = threadIdx.x;
  __shared__ float s1[256], s2[256];
  float v = h[(size_t)t * 256 + j] + r[(size_t)t * 256 + j];
  s1[j] = v; s2[j] = v * v;
  __syncthreads();
  for (int st = 128; st > 0; st >>= 1) {
    if (j < st) { s1[j] += s1[j + st]; s2[j] += s2[j + st]; }
    __syncthreads();
  }
  float mean = s1[0] * (1.0f / 256.0f);
  float var  = s2[0] * (1.0f / 256.0f) - mean * mean;
  float inv  = rsqrtf(var + 1e-5f);
  float o = (v - mean) * inv * g[j] + be[j];
  h[(size_t)t * 256 + j] = o;
  h16[(size_t)t * 256 + j] = (_Float16)o;
}

// ---------------------------------------------------------------------------
// Depthwise causal conv (k=4) + SiLU; dual output (f16 feeds x_proj GEMM).
// ---------------------------------------------------------------------------
__global__ void k_conv_silu(const float* __restrict__ xz, const float* __restrict__ cw,
                            const float* __restrict__ cb, float* __restrict__ xc,
                            _Float16* __restrict__ xc16) {
  int id = blockIdx.x * blockDim.x + threadIdx.x;
  if (id >= NBATCH * DINNER) return;
  int b = id / DINNER, ch = id % DINNER;
  float w[4];
#pragma unroll
  for (int k = 0; k < 4; ++k) w[k] = cw[ch * 4 + k];
  float bias = cb[ch];
  for (int s = 0; s < SEQ; ++s) {
    float acc = bias;
#pragma unroll
    for (int k = 0; k < 4; ++k) {
      int sp = s + k - 3;
      if (sp >= 0) acc += w[k] * xz[(size_t)(b * SEQ + sp) * 1024 + ch];
    }
    float v = silu_f(acc);
    size_t idx = (size_t)(b * SEQ + s) * DINNER + ch;
    xc[idx] = v;
    xc16[idx] = (_Float16)v;
  }
}

// ---------------------------------------------------------------------------
// dt[t][ch] = softplus(sum_r xdb[t][r] * dt_proj_w[ch][r] + dt_proj_b[ch]); K=16.
// ---------------------------------------------------------------------------
__global__ void k_dt(const float* __restrict__ xdb, const float* __restrict__ dpw,
                     const float* __restrict__ dpb, float* __restrict__ dt) {
  int id = blockIdx.x * blockDim.x + threadIdx.x;
  if (id >= T_TOKENS * DINNER) return;
  int t = id / DINNER, ch = id % DINNER;
  const float* xr = xdb + (size_t)t * 48;
  const float* wr = dpw + ch * 16;
  float acc = dpb[ch];
#pragma unroll
  for (int r = 0; r < 16; ++r) acc += xr[r] * wr[r];
  dt[id] = (acc > 20.0f) ? acc : log1pf(expf(acc));
}

// ---------------------------------------------------------------------------
// Selective scan over S=9; dual output (f16 feeds out_proj GEMM).
// ---------------------------------------------------------------------------
__global__ void k_scan(const float* __restrict__ dt, const float* __restrict__ xdb,
                       const float* __restrict__ xc, const float* __restrict__ xz,
                       const float* __restrict__ A_log, const float* __restrict__ D,
                       float* __restrict__ y, _Float16* __restrict__ y16) {
  int id = blockIdx.x * blockDim.x + threadIdx.x;
  if (id >= NBATCH * DINNER) return;
  int b = id / DINNER, ch = id % DINNER;
  float Av[NSTATE], st[NSTATE];
#pragma unroll
  for (int n = 0; n < NSTATE; ++n) { Av[n] = -expf(A_log[ch * NSTATE + n]); st[n] = 0.0f; }
  float Dch = D[ch];
  for (int s = 0; s < SEQ; ++s) {
    size_t t = (size_t)b * SEQ + s;
    float dtv = dt[t * DINNER + ch];
    float xv  = xc[t * DINNER + ch];
    const float* xr = xdb + t * 48;
    float yv = 0.0f;
#pragma unroll
    for (int n = 0; n < NSTATE; ++n) {
      st[n] = st[n] * expf(dtv * Av[n]) + dtv * xr[16 + n] * xv;
      yv += st[n] * xr[32 + n];
    }
    float zv = xz[t * 1024 + 512 + ch];
    float o = (yv + Dch * xv) * silu_f(zv);
    y[t * DINNER + ch] = o;
    y16[t * DINNER + ch] = (_Float16)o;
  }
}

// ---------------------------------------------------------------------------
// o2[t] = o1[t] . w2b + b2b  (K=32)
// ---------------------------------------------------------------------------
__global__ void k_o2(const float* __restrict__ o1, const float* __restrict__ w2b,
                     const float* __restrict__ b2b, float* __restrict__ o2) {
  int t = blockIdx.x * blockDim.x + threadIdx.x;
  if (t >= T_TOKENS) return;
  float acc = b2b[0];
  const float* r = o1 + (size_t)t * 32;
#pragma unroll
  for (int j = 0; j < 32; ++j) acc += r[j] * w2b[j];
  o2[t] = acc;
}

// ---------------------------------------------------------------------------
// out[b] = relu(o2[b,:9] @ w3a^T + b3a) @ w3b^T + b3b
// ---------------------------------------------------------------------------
__global__ void k_head(const float* __restrict__ o2, const float* __restrict__ w3a,
                       const float* __restrict__ b3a, const float* __restrict__ w3b,
                       const float* __restrict__ b3b, float* __restrict__ out) {
  int b = blockIdx.x * blockDim.x + threadIdx.x;
  if (b >= NBATCH) return;
  float o[SEQ];
#pragma unroll
  for (int i = 0; i < SEQ; ++i) o[i] = o2[b * SEQ + i];
  float acc = b3b[0];
#pragma unroll
  for (int j = 0; j < SEQ; ++j) {
    float r = b3a[j];
#pragma unroll
    for (int i = 0; i < SEQ; ++i) r += o[i] * w3a[j * SEQ + i];
    acc += fmaxf(r, 0.0f) * w3b[j];
  }
  out[b] = acc;
}

// ---------------------------------------------------------------------------
extern "C" void kernel_launch(void* const* d_in, const int* in_sizes, int n_in,
                              void* d_out, int out_size, void* d_ws, size_t ws_size,
                              hipStream_t stream) {
  (void)in_sizes; (void)n_in; (void)out_size; (void)ws_size;
  const float* x          = (const float*)d_in[0];
  const float* w1         = (const float*)d_in[1];
  const float* b1         = (const float*)d_in[2];
  const float* attn_in_w  = (const float*)d_in[3];   // (2,768,256)
  const float* attn_in_b  = (const float*)d_in[4];
  const float* attn_out_w = (const float*)d_in[5];   // (2,256,256)
  const float* attn_out_b = (const float*)d_in[6];
  const float* ln1_g      = (const float*)d_in[7];
  const float* ln1_b      = (const float*)d_in[8];
  const float* ffw_w1     = (const float*)d_in[9];   // (2,256,256)
  const float* ffw_b1     = (const float*)d_in[10];
  const float* ffw_w2     = (const float*)d_in[11];
  const float* ffw_b2     = (const float*)d_in[12];
  const float* ln2_g      = (const float*)d_in[13];
  const float* ln2_b      = (const float*)d_in[14];
  const float* in_proj_w  = (const float*)d_in[15];  // (1024,256)
  const float* conv_w     = (const float*)d_in[16];  // (512,4)
  const float* conv_b     = (const float*)d_in[17];
  const float* x_proj_w   = (const float*)d_in[18];  // (48,512)
  const float* dt_proj_w  = (const float*)d_in[19];  // (512,16)
  const float* dt_proj_b  = (const float*)d_in[20];
  const float* A_log      = (const float*)d_in[21];  // (512,16)
  const float* Dp         = (const float*)d_in[22];
  const float* out_proj_w = (const float*)d_in[23];  // (256,512)
  const float* w2a        = (const float*)d_in[24];  // (32,256)
  const float* b2a        = (const float*)d_in[25];
  const float* w2b        = (const float*)d_in[26];
  const float* b2b        = (const float*)d_in[27];
  const float* w3a        = (const float*)d_in[28];
  const float* b3a        = (const float*)d_in[29];
  const float* w3b        = (const float*)d_in[30];
  const float* b3b        = (const float*)d_in[31];
  float* out = (float*)d_out;

  const size_t T = T_TOKENS;
  float* ws      = (float*)d_ws;
  // ---- fp32 buffers ----
  float* buf_h   = ws;                   // T*256  (h, later h2)
  float* buf_big = buf_h   + T * 256;    // T*1024 (qkv, later xz)
  float* buf_t   = buf_big + T * 1024;   // T*256  (attn out / ffw out)
  float* buf_f1  = buf_t   + T * 256;    // T*256  (proj tmp / ffw hidden / o1)
  float* buf_xc  = buf_f1  + T * 256;    // T*512
  float* buf_dt  = buf_xc  + T * 512;    // T*512
  float* buf_y   = buf_dt  + T * 512;    // T*512
  float* buf_xdb = buf_y   + T * 512;    // T*48
  float* buf_o2  = buf_xdb + T * 48;     // T
  // ---- f16 arenas (after fp32 region) ----
  _Float16* h16f = (_Float16*)(buf_o2 + T);
  _Float16* w_attn_in16  = h16f;                                 // 2*768*256
  _Float16* w_attn_out16 = w_attn_in16  + 2 * 768 * 256;         // 2*256*256
  _Float16* w_ffw1_16    = w_attn_out16 + 2 * 256 * 256;         // 2*256*256
  _Float16* w_ffw2_16    = w_ffw1_16    + 2 * 256 * 256;         // 2*256*256
  _Float16* w_inproj16   = w_ffw2_16    + 2 * 256 * 256;         // 1024*256
  _Float16* w_xproj16    = w_inproj16   + 1024 * 256;            // 48*512
  _Float16* w_outproj16  = w_xproj16    + 48 * 512;              // 256*512
  _Float16* w_w2a16      = w_outproj16  + 256 * 512;             // 32*256
  _Float16* h16  = w_w2a16 + 32 * 256;   // T*256  (h / h2 mirror)
  _Float16* s16  = h16 + T * 256;        // T*256  (attn out / ffw hidden mirror)
  _Float16* xc16 = s16 + T * 256;        // T*512
  _Float16* y16  = xc16 + T * 512;       // T*512

  auto cvt = [&](const float* src, _Float16* dst, int n) {
    k_cvt16<<<(n + 255) / 256, 256, 0, stream>>>(src, dst, n);
  };
  auto gemm = [&](const _Float16* A16, const _Float16* W16, const float* bias,
                  float* C, _Float16* C16, int N, int K, int relu) {
    if (N % 64 == 0) {
      int groups = (T_TOKENS / 16) * (N / 64);
      k_gemm_wmma_f16<4><<<(groups + 7) / 8, 256, 0, stream>>>(
          A16, W16, bias, C, C16, T_TOKENS, N, K, relu);
    } else {
      int groups = (T_TOKENS / 16) * (N / 16);
      k_gemm_wmma_f16<1><<<(groups + 7) / 8, 256, 0, stream>>>(
          A16, W16, bias, C, C16, T_TOKENS, N, K, relu);
    }
  };

  // ---- one-time weight conversion to f16 ----
  cvt(attn_in_w,  w_attn_in16,  2 * 768 * 256);
  cvt(attn_out_w, w_attn_out16, 2 * 256 * 256);
  cvt(ffw_w1,     w_ffw1_16,    2 * 256 * 256);
  cvt(ffw_w2,     w_ffw2_16,    2 * 256 * 256);
  cvt(in_proj_w,  w_inproj16,   1024 * 256);
  cvt(x_proj_w,   w_xproj16,    48 * 512);
  cvt(out_proj_w, w_outproj16,  256 * 512);
  cvt(w2a,        w_w2a16,      32 * 256);

  // h = x @ w1^T + b1
  k_embed<<<(T_TOKENS * DMODEL + 255) / 256, 256, 0, stream>>>(x, w1, b1, buf_h, h16);

  const int attn_lds = (3 * SEQ * 128 + SEQ * SEQ) * sizeof(float);  // 14148 B

  for (int i = 0; i < 2; ++i) {
    // qkv = h @ attn_in_w^T + b
    gemm(h16, w_attn_in16 + (size_t)i * 768 * 256, attn_in_b + i * 768,
         buf_big, nullptr, 768, 256, 0);
    // attention (per batch, per head) with TDM-staged q/k/v tiles
    k_attn<<<dim3(NBATCH, 2), 128, attn_lds, stream>>>(buf_big, buf_t, s16);
    // out proj
    gemm(s16, w_attn_out16 + (size_t)i * 256 * 256, attn_out_b + i * 256,
         buf_f1, nullptr, 256, 256, 0);
    // h = LN(h + proj)
    k_add_ln<<<T_TOKENS, 256, 0, stream>>>(buf_h, buf_f1, ln1_g + i * 256, ln1_b + i * 256, h16);
    // ffw
    gemm(h16, w_ffw1_16 + (size_t)i * 256 * 256, ffw_b1 + i * 256,
         buf_f1, s16, 256, 256, 1);
    gemm(s16, w_ffw2_16 + (size_t)i * 256 * 256, ffw_b2 + i * 256,
         buf_t, nullptr, 256, 256, 0);
    // h = LN(h + f)
    k_add_ln<<<T_TOKENS, 256, 0, stream>>>(buf_h, buf_t, ln2_g + i * 256, ln2_b + i * 256, h16);
  }

  // xz = h @ in_proj_w^T
  gemm(h16, w_inproj16, nullptr, buf_big, nullptr, 1024, 256, 0);
  // xc = silu(causal_conv(xm))
  k_conv_silu<<<(NBATCH * DINNER + 255) / 256, 256, 0, stream>>>(buf_big, conv_w, conv_b,
                                                                 buf_xc, xc16);
  // xdb = xc @ x_proj_w^T   (N=48 -> TN=1 path)
  gemm(xc16, w_xproj16, nullptr, buf_xdb, nullptr, 48, 512, 0);
  // dt = softplus(dti @ dt_proj_w^T + dt_proj_b)
  k_dt<<<(T_TOKENS * DINNER + 255) / 256, 256, 0, stream>>>(buf_xdb, dt_proj_w, dt_proj_b, buf_dt);
  // selective scan + D skip + z gating
  k_scan<<<(NBATCH * DINNER + 255) / 256, 256, 0, stream>>>(buf_dt, buf_xdb, buf_xc, buf_big,
                                                            A_log, Dp, buf_y, y16);
  // h2 = y @ out_proj_w^T (dual write: h16 feeds next GEMM)
  gemm(y16, w_outproj16, nullptr, buf_h, h16, 256, 512, 0);
  // o1 = h2 @ w2a^T + b2a  (N=32 -> TN=1 path)
  gemm(h16, w_w2a16, b2a, buf_f1, nullptr, 32, 256, 0);
  // o2[t] = o1 . w2b + b2b
  k_o2<<<(T_TOKENS + 255) / 256, 256, 0, stream>>>(buf_f1, w2b, b2b, buf_o2);
  // out[b] = relu(o2 @ w3a^T + b3a) @ w3b^T + b3b
  k_head<<<(NBATCH + 255) / 256, 256, 0, stream>>>(buf_o2, w3a, b3a, w3b, b3b, out);
}
